// APPNP_88175678587122
// MI455X (gfx1250) — matrix operand order repaired and checked by reading, not compiled, not run on hardware
//
#include <hip/hip_runtime.h>
#include <hip/hip_bf16.h>
#include <stdint.h>

// ---------------------------------------------------------------------------
// APPNP for MI455X (gfx1250, wave32).
//  - MLP uses v_wmma_f32_16x16x32_bf16; all fragment traffic is b128
//    (global_load_b128 for X, ds_load_b128 for W1^T/W2^T/H tiles).
//  - Propagation: wave-per-edge, 32-edge batches via shuffles, float2 gathers
//    (Z is L2-resident: 25.6MB << 192MB L2), fp32 global atomics for scatter.
// ---------------------------------------------------------------------------

typedef __attribute__((ext_vector_type(16))) __bf16 v16bf;
typedef __attribute__((ext_vector_type(8)))  __bf16 v8bf;
typedef __attribute__((ext_vector_type(8)))  float  v8f;

#define IN_DIM 512
#define HID    64
#define OUTD   64
#define ALPHA  0.1f
#define KEEP_TH  3865470566u      /* 0.9 * 2^32 */
#define INV_KEEP (1.0f / 0.9f)

#define W1_PAD 40                 /* 80B rows: 16B-aligned v8bf runs */
#define W2_PAD 72                 /* 144B rows */
#define H_PAD  72

__device__ __forceinline__ unsigned hash32(unsigned x) {
    x ^= x >> 16; x *= 0x7feb352dU;
    x ^= x >> 15; x *= 0x846ca68bU;
    x ^= x >> 16; return x;
}

__device__ __forceinline__ __bf16 f2bf(float f) {
    unsigned u = __builtin_bit_cast(unsigned, f);
    u += 0x7fffU + ((u >> 16) & 1u);            // round-to-nearest-even
    unsigned short s = (unsigned short)(u >> 16);
    return __builtin_bit_cast(__bf16, s);
}

__device__ __forceinline__ v16bf cat16(v8bf lo, v8bf hi) {
    return __builtin_shufflevector(lo, hi, 0, 1, 2, 3, 4, 5, 6, 7,
                                           8, 9, 10, 11, 12, 13, 14, 15);
}

// ---------------------------------------------------------------------------
// MLP: Z0 = dropout(relu(dropout(X) @ W1 + b1)) @ W2 + b2
// Block = 256 threads = 8 waves; each wave computes a 16x64 output tile.
// ---------------------------------------------------------------------------
__global__ __launch_bounds__(256) void appnp_mlp_kernel(
    const float* __restrict__ X,  const float* __restrict__ W1,
    const float* __restrict__ b1, const float* __restrict__ W2,
    const float* __restrict__ b2, float* __restrict__ Z0, int N)
{
    __shared__ __bf16 sW1t[64][W1_PAD];      // W1^T k-slice: [n][k_local 0..31]
    __shared__ __bf16 sW2t[64][W2_PAD];      // W2^T full:    [n][k 0..63]
    __shared__ __bf16 sHt[8][16][H_PAD];     // per-wave hidden tile [m][col]

    const int tid  = threadIdx.x;
    const int wave = tid >> 5;
    const int lane = tid & 31;
    const int hs   = lane >> 4;              // half-wave select
    const int ml   = lane & 15;
    const int rowbase = blockIdx.x * 128 + wave * 16;

    // Stage W2^T (64x64) once: float4 reads, bf16 scatter into LDS.
    #pragma unroll
    for (int i = 0; i < 4; ++i) {
        int quad = tid + i * 256;            // 1024 quads total
        int k = quad >> 4; int nq = (quad & 15) * 4;
        float4 w = *(const float4*)(W2 + k * OUTD + nq);
        sW2t[nq + 0][k] = f2bf(w.x);
        sW2t[nq + 1][k] = f2bf(w.y);
        sW2t[nq + 2][k] = f2bf(w.z);
        sW2t[nq + 3][k] = f2bf(w.w);
    }

    const v8f vzero = {0.f, 0.f, 0.f, 0.f, 0.f, 0.f, 0.f, 0.f};
    v8f acc[4];
    #pragma unroll
    for (int t = 0; t < 4; ++t) acc[t] = vzero;

    const int  rowg  = rowbase + ml;
    const bool rowok = rowg < N;
    const float* xrow = X + (size_t)rowg * IN_DIM;

    // ---- GEMM1: (16 x 512) @ (512 x 64), K in steps of 32 ----
    for (int kk = 0; kk < IN_DIM; kk += 32) {
        __syncthreads();
        // Stage W1^T slice [64 n][32 k]: float4 reads, bf16 scatter.
        #pragma unroll
        for (int i = 0; i < 2; ++i) {
            int quad = tid + i * 256;        // 512 quads total
            int k = quad >> 4; int nq = (quad & 15) * 4;
            float4 w = *(const float4*)(W1 + (size_t)(kk + k) * HID + nq);
            sW1t[nq + 0][k] = f2bf(w.x);
            sW1t[nq + 1][k] = f2bf(w.y);
            sW1t[nq + 2][k] = f2bf(w.z);
            sW1t[nq + 3][k] = f2bf(w.w);
        }
        // Prefetch next X slice while this k-step computes.
        if (rowok && (kk + 32) < IN_DIM)
            __builtin_prefetch(xrow + kk + 32 + 8 * hs, 0, 3);
        __syncthreads();

        // A fragment: row ml, K runs {8*hs..+7} and {16+8*hs..+7}
        // Both runs 32B-aligned -> global_load_b128 x4.
        const int kA0 = kk + 8 * hs;
        const int kA1 = kk + 16 + 8 * hs;
        float4 xa0, xa1, xb0, xb1;
        if (rowok) {
            xa0 = *(const float4*)(xrow + kA0);
            xa1 = *(const float4*)(xrow + kA0 + 4);
            xb0 = *(const float4*)(xrow + kA1);
            xb1 = *(const float4*)(xrow + kA1 + 4);
        } else {
            xa0 = xa1 = xb0 = xb1 = make_float4(0.f, 0.f, 0.f, 0.f);
        }
        float xs[16] = {xa0.x, xa0.y, xa0.z, xa0.w, xa1.x, xa1.y, xa1.z, xa1.w,
                        xb0.x, xb0.y, xb0.z, xb0.w, xb1.x, xb1.y, xb1.z, xb1.w};
        v16bf a;
        #pragma unroll
        for (int e = 0; e < 16; ++e) {
            int kglob = (e < 8) ? (kA0 + e) : (kA1 + e - 8);
            unsigned ih = 0x9E3779B1u ^ (unsigned)(rowg * IN_DIM + kglob);
            float x = (hash32(ih) < KEEP_TH) ? xs[e] * INV_KEEP : 0.f;
            a[e] = f2bf(x);
        }

        #pragma unroll
        for (int nt = 0; nt < 4; ++nt) {
            const __bf16* bp = &sW1t[nt * 16 + ml][8 * hs];    // 16B-aligned
            v16bf bfr = cat16(*(const v8bf*)bp, *(const v8bf*)(bp + 16));
            acc[nt] = __builtin_amdgcn_wmma_f32_16x16x32_bf16(
                false, a, false, bfr, (short)0, acc[nt], false, false);
        }
    }

    // ---- epilogue 1: bias + relu + dropout -> bf16 hidden tile in LDS ----
    __syncthreads();
    #pragma unroll
    for (int nt = 0; nt < 4; ++nt) {
        int col = nt * 16 + ml;
        float bias = b1[col];
        #pragma unroll
        for (int r = 0; r < 8; ++r) {
            int m  = r + 8 * hs;               // C layout: vgpr r -> row r(+8)
            int rg = rowbase + m;
            float h = acc[nt][r] + bias;
            h = h > 0.f ? h : 0.f;
            unsigned ih = 0x85EBCA77u ^ (unsigned)(rg * HID + col);
            h = (hash32(ih) < KEEP_TH) ? h * INV_KEEP : 0.f;
            sHt[wave][m][col] = f2bf(h);
        }
    }
    __syncthreads();

    // ---- GEMM2: (16 x 64) @ (64 x 64) ----
    v8f zac[4];
    #pragma unroll
    for (int t = 0; t < 4; ++t) zac[t] = vzero;

    #pragma unroll
    for (int kk = 0; kk < HID; kk += 32) {
        const __bf16* ap0 = &sHt[wave][ml][kk + 8 * hs];        // 16B-aligned
        const __bf16* ap1 = &sHt[wave][ml][kk + 16 + 8 * hs];
        v16bf a2 = cat16(*(const v8bf*)ap0, *(const v8bf*)ap1);
        #pragma unroll
        for (int nt = 0; nt < 4; ++nt) {
            const __bf16* bp = &sW2t[nt * 16 + ml][kk + 8 * hs];
            v16bf bfr = cat16(*(const v8bf*)bp, *(const v8bf*)(bp + 16));
            zac[nt] = __builtin_amdgcn_wmma_f32_16x16x32_bf16(
                false, a2, false, bfr, (short)0, zac[nt], false, false);
        }
    }

    // ---- epilogue 2: bias, store Z0 ----
    #pragma unroll
    for (int nt = 0; nt < 4; ++nt) {
        int col = nt * 16 + ml;
        float bias = b2[col];
        #pragma unroll
        for (int r = 0; r < 8; ++r) {
            int m  = r + 8 * hs;
            int rg = rowbase + m;
            if (rg < N) Z0[(size_t)rg * OUTD + col] = zac[nt][r] + bias;
        }
    }
}

// ---------------------------------------------------------------------------
// Per-hop init: Zout = ALPHA * Z0  (vectorized float4)
// ---------------------------------------------------------------------------
__global__ __launch_bounds__(256) void appnp_init_kernel(
    const float* __restrict__ Z0, float* __restrict__ Zout, int n4)
{
    int i = blockIdx.x * blockDim.x + threadIdx.x;
    if (i < n4) {
        float4 z = ((const float4*)Z0)[i];
        float4 o;
        o.x = ALPHA * z.x; o.y = ALPHA * z.y;
        o.z = ALPHA * z.z; o.w = ALPHA * z.w;
        ((float4*)Zout)[i] = o;
    }
}

// ---------------------------------------------------------------------------
// Propagation hop: Zout[row] += w_e * Zin[col] over all edges.
// One wave per edge; 32-edge batches; lanes cover 64 features via float2.
// ---------------------------------------------------------------------------
__global__ __launch_bounds__(256) void appnp_prop_kernel(
    const int* __restrict__ rows, const int* __restrict__ cols,
    const float* __restrict__ value, const float* __restrict__ Zin,
    float* __restrict__ Zout, int E, unsigned seed)
{
    const int lane = threadIdx.x & 31;
    const int gw   = blockIdx.x * (blockDim.x >> 5) + (threadIdx.x >> 5);
    const int nw   = gridDim.x * (blockDim.x >> 5);

    for (int base = gw * 32; base < E; base += nw * 32) {
        int e = base + lane;
        int r = 0, c = 0; float w = 0.f;
        if (e < E) {
            r = rows[e];
            c = cols[e];
            float v = value[e];
            unsigned h = hash32(seed + (unsigned)e * 0x9E3779B9u);
            // (1-ALPHA) * (v / keep); dropped edges -> 0 (skip atomics)
            w = (h < KEEP_TH) ? v * ((1.0f - ALPHA) * INV_KEEP) : 0.f;
        }
        #pragma unroll 4
        for (int i = 0; i < 32; ++i) {
            float wi = __shfl(w, i, 32);
            if (wi != 0.f) {                 // wave-uniform branch
                int ci = __shfl(c, i, 32);
                int ri = __shfl(r, i, 32);
                float2 z = *(const float2*)(Zin + (size_t)ci * OUTD + lane * 2);
                float* dst = Zout + (size_t)ri * OUTD + lane * 2;
                atomicAdd(dst,     z.x * wi);
                atomicAdd(dst + 1, z.y * wi);
            }
        }
    }
}

// ---------------------------------------------------------------------------
extern "C" void kernel_launch(void* const* d_in, const int* in_sizes, int n_in,
                              void* d_out, int out_size, void* d_ws, size_t ws_size,
                              hipStream_t stream)
{
    const int*   index = (const int*)d_in[0];     // [2, E]: rows then cols
    const float* value = (const float*)d_in[1];   // [E]
    const float* X     = (const float*)d_in[3];   // [N, 512]
    const float* W1    = (const float*)d_in[4];
    const float* b1    = (const float*)d_in[5];
    const float* W2    = (const float*)d_in[6];
    const float* b2    = (const float*)d_in[7];

    const int E = in_sizes[1];
    const int N = in_sizes[3] / IN_DIM;

    float* Z0   = (float*)d_ws;                   // [N,64]
    float* bufA = Z0 + (size_t)N * OUTD;          // [N,64] ping buffer
    float* out  = (float*)d_out;                  // [N,64] final

    // MLP -> Z0
    appnp_mlp_kernel<<<(N + 127) / 128, 256, 0, stream>>>(X, W1, b1, W2, b2, Z0, N);

    const int n4 = (N * OUTD) / 4;
    for (int h = 0; h < 10; ++h) {
        // ping-pong so hop 9 (last) lands in d_out
        const float* Zin  = (h == 0) ? Z0 : ((h & 1) ? bufA : out);
        float*       Zout = (h & 1) ? out : bufA;
        appnp_init_kernel<<<(n4 + 255) / 256, 256, 0, stream>>>(Z0, Zout, n4);
        appnp_prop_kernel<<<8192, 256, 0, stream>>>(
            index, index + E, value, Zin, Zout, E,
            0xC2B2AE35u + 0x9E3779B9u * (unsigned)h);
    }
}
